// DualAttentionNetwork_3092376453893
// MI455X (gfx1250) — matrix operand order repaired and checked by reading, not compiled, not run on hardware
//
#include <hip/hip_runtime.h>
#include <hip/hip_bf16.h>

typedef __attribute__((ext_vector_type(16))) _Float16 v16h;
typedef __attribute__((ext_vector_type(8)))  _Float16 v8h;
typedef __attribute__((ext_vector_type(8)))  float    v8f;

#define BB 32
#define NN 2000
#define MM 512
#define NM 2512
#define HO 128
#define ALPHA 0.2f
#define NEGV  -9e15f

__device__ __forceinline__ v8f wmma16(v16h a, v16h b, v8f c) {
  return __builtin_amdgcn_wmma_f32_16x16x32_f16(false, a, false, b, (short)0, c,
                                                false, false);
}
__device__ __forceinline__ float lrelu(float x) { return x > 0.f ? x : ALPHA * x; }
__device__ __forceinline__ float eluf(float x)  { return x > 0.f ? x : (__expf(x) - 1.f); }

// A fragment: rows in lane r, K map = {g*8+j (j<8), g*8+16+j-8 (j>=8)} per
// K-chunk => two contiguous 8-float runs -> 4x float4 loads.
__device__ __forceinline__ v16h load_a_frag(const float* fj, int kc, int g) {
  const float4* f4 = (const float4*)fj;
  const int b4 = kc * 8 + g * 2;
  const float4 p0 = f4[b4], p1 = f4[b4 + 1], p2 = f4[b4 + 4], p3 = f4[b4 + 5];
  v16h a;
  a[0] = (_Float16)p0.x;  a[1] = (_Float16)p0.y;  a[2] = (_Float16)p0.z;  a[3] = (_Float16)p0.w;
  a[4] = (_Float16)p1.x;  a[5] = (_Float16)p1.y;  a[6] = (_Float16)p1.z;  a[7] = (_Float16)p1.w;
  a[8] = (_Float16)p2.x;  a[9] = (_Float16)p2.y;  a[10] = (_Float16)p2.z; a[11] = (_Float16)p2.w;
  a[12] = (_Float16)p3.x; a[13] = (_Float16)p3.y; a[14] = (_Float16)p3.z; a[15] = (_Float16)p3.w;
  return a;
}

// B fragment from a pre-packed fragment line (16 f16 contiguous): 2x b128.
__device__ __forceinline__ v16h load_line(const _Float16* line) {
  const v8h* p = (const v8h*)line;
  const v8h lo = p[0], hi = p[1];
  return __builtin_shufflevector(lo, hi, 0, 1, 2, 3, 4, 5, 6, 7, 8, 9, 10, 11, 12, 13, 14, 15);
}

// ---------------- setup: pack W_op / W_m into WMMA B-fragment order ----------------
// layout: [(kc*8 + t)*32 + lane]*16 + j ;  value = W[(h*64 + kc*32 + g*8 + j + (j&8))*32 + o]
__global__ void __launch_bounds__(256)
pack_w(const float* __restrict__ W_op, const float* __restrict__ W_m,
       _Float16* __restrict__ wpack_op, _Float16* __restrict__ wpack_m) {
  for (int i = threadIdx.x; i < 8192; i += 256) {
    const int j = i & 15, lane = (i >> 4) & 31, t = (i >> 9) & 7, kc = i >> 12;
    const int col = t * 16 + (lane & 15), h = col >> 5, o = col & 31, g = lane >> 4;
    const int K = kc * 32 + g * 8 + j + (j & 8);
    wpack_op[i] = (_Float16)W_op[(h * 64 + K) * 32 + o];
    wpack_m[i]  = (_Float16)W_m[(h * 64 + K) * 32 + o];
  }
}

// ---------------- op projection: Wh = fea_j @ W_op(concat), full rows to ws ----------------
__global__ void __launch_bounds__(32)
op_proj(const float* __restrict__ fea_j, const _Float16* __restrict__ wpack,
        const float* __restrict__ a_op, float* __restrict__ whj,
        float* __restrict__ wh1j, float* __restrict__ wh2j) {
  const int n0 = blockIdx.x * 16, b = blockIdx.y;
  const int lane = threadIdx.x, g = lane >> 4, r = lane & 15;
  __shared__ float wh[16][HO];

  const float* fj = fea_j + ((size_t)b * NN + n0 + r) * 64;
  const v16h a0 = load_a_frag(fj, 0, g), a1 = load_a_frag(fj, 1, g);
#pragma unroll
  for (int t = 0; t < 8; ++t) {
    v8f c = {};
    c = wmma16(a0, load_line(wpack + ((0 * 8 + t) * 32 + lane) * 16), c);
    c = wmma16(a1, load_line(wpack + ((1 * 8 + t) * 32 + lane) * 16), c);
#pragma unroll
    for (int v = 0; v < 8; ++v) wh[g * 8 + v][t * 16 + r] = c[v];
  }
  __syncthreads();

  // coalesced tile store (float4)
  float4* dst4 = (float4*)(whj + ((size_t)b * NN + n0) * HO);
  const float4* src4 = (const float4*)&wh[0][0];
  for (int i = lane; i < 16 * HO / 4; i += 32) dst4[i] = src4[i];

  // Wh1/Wh2 row dots
#pragma unroll
  for (int it = 0; it < 4; ++it) {
    const int task = lane + 32 * it;     // 0..127
    const int which = task >> 6, t2 = task & 63, rr = t2 >> 2, h = t2 & 3;
    const float* av = a_op + h * 64 + which * 32;
    float s = 0.f;
    for (int o = 0; o < 32; ++o) s += wh[rr][h * 32 + o] * av[o];
    (which ? wh2j : wh1j)[((size_t)b * 4 + h) * NN + n0 + rr] = s;
  }
}

// ---------------- op attention: 3-neighbor softmax + combine + elu ----------------
__global__ void __launch_bounds__(32)
op_attn2(const int* __restrict__ op_mask, const float* __restrict__ whj,
         const float* __restrict__ wh1j, const float* __restrict__ wh2j,
         float* __restrict__ out) {
  const int n0 = blockIdx.x * 16, b = blockIdx.y;
  const int lane = threadIdx.x;
#pragma unroll
  for (int it = 0; it < 2; ++it) {
    const int task = lane + 32 * it, rr = task >> 2, h = task & 3;
    const int n = n0 + rr;
    const int np = (n + NN - 1) % NN, nx = (n + 1) % NN;
    const float w1 = wh1j[((size_t)b * 4 + h) * NN + n];
    const float* w2b = wh2j + ((size_t)b * 4 + h) * NN;
    const float w2[3] = {w2b[np], w2b[n], w2b[nx]};
    float e[3], mx = -INFINITY;
#pragma unroll
    for (int k = 0; k < 3; ++k) {
      float v = lrelu(w1 + w2[k]);
      if (op_mask[((size_t)b * NN + n) * 3 + k] > 0) v = NEGV;  // op block: mask>0 -> NEG
      e[k] = v;
      mx = fmaxf(mx, v);
    }
    float s = 0.f;
#pragma unroll
    for (int k = 0; k < 3; ++k) { e[k] = __expf(e[k] - mx); s += e[k]; }
    const float inv = 1.f / s;
    const float4* p0 = (const float4*)(whj + ((size_t)b * NN + np) * HO + h * 32);
    const float4* p1 = (const float4*)(whj + ((size_t)b * NN + n) * HO + h * 32);
    const float4* p2 = (const float4*)(whj + ((size_t)b * NN + nx) * HO + h * 32);
    float4* op4 = (float4*)(out + ((size_t)b * NM + n) * HO + h * 32);
#pragma unroll
    for (int q = 0; q < 8; ++q) {
      const float4 x0 = p0[q], x1 = p1[q], x2 = p2[q];
      float4 y;
      y.x = eluf((e[0] * x0.x + e[1] * x1.x + e[2] * x2.x) * inv);
      y.y = eluf((e[0] * x0.y + e[1] * x1.y + e[2] * x2.y) * inv);
      y.z = eluf((e[0] * x0.z + e[1] * x1.z + e[2] * x2.z) * inv);
      y.w = eluf((e[0] * x0.w + e[1] * x1.w + e[2] * x2.w) * inv);
      op4[q] = y;
    }
  }
}

// ---------------- machine projection: Whm via WMMA, emit fragment-ordered f16 ----------------
// whmfrag layout per (b,h): [(kc*2 + t)*32 + lane]*16 + j  (16384 halves)
__global__ void __launch_bounds__(32)
mch_proj(const float* __restrict__ fea_m, const _Float16* __restrict__ wpack,
         const float* __restrict__ a_m, _Float16* __restrict__ whmfrag,
         float* __restrict__ ws_wh1, float* __restrict__ ws_wh2) {
  const int m0 = blockIdx.x * 16, b = blockIdx.y;
  const int lane = threadIdx.x, g = lane >> 4, r = lane & 15;
  __shared__ float wh[16][HO];

  const float* fm = fea_m + ((size_t)b * MM + m0 + r) * 64;
  const v16h a0 = load_a_frag(fm, 0, g), a1 = load_a_frag(fm, 1, g);
#pragma unroll
  for (int t = 0; t < 8; ++t) {
    v8f c = {};
    c = wmma16(a0, load_line(wpack + ((0 * 8 + t) * 32 + lane) * 16), c);
    c = wmma16(a1, load_line(wpack + ((1 * 8 + t) * 32 + lane) * 16), c);
#pragma unroll
    for (int v = 0; v < 8; ++v) wh[g * 8 + v][t * 16 + r] = c[v];
  }
  __syncthreads();

  // pack this 16-row slab of Whm into B-fragment lines for mch_attn:
  // rows m0+j (j<8) -> lane_g=0, elem gl*8+j; rows m0+8+j -> lane_g=1, elem gl*8+j.
  const int gl = (m0 >> 4) & 1, kcb = m0 >> 5;
#pragma unroll
  for (int i = 0; i < 4; ++i) {
    const int L = lane + 32 * i;                 // 0..127: h(4) x t(2) x n(16)
    const int h = L >> 5, t = (L >> 4) & 1, n = L & 15;
    _Float16* dst =
        whmfrag + (((((size_t)b * 4 + h) * 16 + kcb) * 2 + t) * 32 + n) * 16 + gl * 8;
    v8h lo, hi;
#pragma unroll
    for (int j = 0; j < 8; ++j) lo[j] = (_Float16)wh[j][h * 32 + t * 16 + n];
#pragma unroll
    for (int j = 0; j < 8; ++j) hi[j] = (_Float16)wh[j + 8][h * 32 + t * 16 + n];
    ((v8h*)dst)[0] = lo;
    ((v8h*)(dst + 16 * 16))[0] = hi;
  }

#pragma unroll
  for (int it = 0; it < 4; ++it) {
    const int task = lane + 32 * it;
    const int which = task >> 6, t2 = task & 63, rr = t2 >> 2, h = t2 & 3;
    const float* av = a_m + h * 96 + which * 32;
    float s = 0.f;
    for (int o = 0; o < 32; ++o) s += wh[rr][h * 32 + o] * av[o];
    (which ? ws_wh2 : ws_wh1)[((size_t)b * 4 + h) * MM + m0 + rr] = s;
  }
}

// ---------------- machine attention: single comp_val pass, online softmax ----------------
__global__ void __launch_bounds__(32)
mch_attn(const float* __restrict__ comp_val, const int* __restrict__ mch_mask,
         const float* __restrict__ W_edge, const float* __restrict__ a_m,
         const _Float16* __restrict__ whmfrag, const float* __restrict__ ws_wh1,
         const float* __restrict__ ws_wh2, float* __restrict__ out) {
  const int m0 = blockIdx.x * 16, b = blockIdx.y;
  const int lane = threadIdx.x, g = lane >> 4, r = lane & 15;

  __shared__ float wh2s[4][MM];
  __shared__ float eas[4][8];

  for (int i = lane; i < 4 * MM; i += 32)
    ((float*)wh2s)[i] = ws_wh2[(size_t)b * 4 * MM + i];
  {
    const int h = lane >> 3, e = lane & 7;
    float s = 0.f;
    for (int o = 0; o < 32; ++o) s += W_edge[(h * 8 + e) * 32 + o] * a_m[h * 96 + 64 + o];
    eas[h][e] = s;
  }
  __syncthreads();

  float ea[4][8];
#pragma unroll
  for (int h = 0; h < 4; ++h)
#pragma unroll
    for (int e = 0; e < 8; ++e) ea[h][e] = eas[h][e];

  float wh1r[4];
#pragma unroll
  for (int h = 0; h < 4; ++h) wh1r[h] = ws_wh1[((size_t)b * 4 + h) * MM + m0 + r];

  v8f acc[4][2];
  float run_max[4], run_sum[4];
#pragma unroll
  for (int h = 0; h < 4; ++h) {
    acc[h][0] = (v8f){};
    acc[h][1] = (v8f){};
    run_max[h] = -INFINITY;
    run_sum[h] = 0.f;
  }

  const int myrow = m0 + r;
  const float* cvrow = comp_val + ((size_t)b * MM + myrow) * MM * 8;
  const int* mkrow = mch_mask + ((size_t)b * MM + myrow) * MM;
  const _Float16* wbase = whmfrag + ((size_t)b * 4) * 16 * 2 * 32 * 16;

  for (int kc = 0; kc < 16; ++kc) {
    // prefetch next chunk of comp_val (speculative; OOB silently dropped)
    __builtin_prefetch(cvrow + ((size_t)(kc + 1) * 32 + g * 8) * 8, 0, 3);

    const int q0 = kc * 32 + g * 8;
    // mask: two contiguous 8-int runs -> 4x int4
    const int4* mk4 = (const int4*)(mkrow + q0);
    const int4 ma = mk4[0], mb = mk4[1], mc = mk4[4], md = mk4[5];
    unsigned msk = 0;
    msk |= (ma.x > 0) << 0;  msk |= (ma.y > 0) << 1;  msk |= (ma.z > 0) << 2;  msk |= (ma.w > 0) << 3;
    msk |= (mb.x > 0) << 4;  msk |= (mb.y > 0) << 5;  msk |= (mb.z > 0) << 6;  msk |= (mb.w > 0) << 7;
    msk |= (mc.x > 0) << 8;  msk |= (mc.y > 0) << 9;  msk |= (mc.z > 0) << 10; msk |= (mc.w > 0) << 11;
    msk |= (md.x > 0) << 12; msk |= (md.y > 0) << 13; msk |= (md.z > 0) << 14; msk |= (md.w > 0) << 15;

    float edh[4][16];
#pragma unroll
    for (int j = 0; j < 16; ++j) {
      const int q = q0 + j + (j & 8);
      const float4* cv4 = (const float4*)(cvrow + (size_t)q * 8);
      const float4 c0 = cv4[0], c1 = cv4[1];
      const float cve[8] = {c0.x, c0.y, c0.z, c0.w, c1.x, c1.y, c1.z, c1.w};
#pragma unroll
      for (int h = 0; h < 4; ++h) {
        float s = 0.f;
#pragma unroll
        for (int e = 0; e < 8; ++e) s += cve[e] * ea[h][e];
        edh[h][j] = s;
      }
    }

#pragma unroll
    for (int h = 0; h < 4; ++h) {
      float sc[16], cmax = -INFINITY;
#pragma unroll
      for (int j = 0; j < 16; ++j) {
        const int q = q0 + j + (j & 8);
        float s = lrelu(wh1r[h] + wh2s[h][q] + edh[h][j]);
        if (!((msk >> j) & 1)) s = NEGV;  // machine block keeps where mask>0
        sc[j] = s;
        cmax = fmaxf(cmax, s);
      }
      cmax = fmaxf(cmax, __shfl_xor(cmax, 16, 32));
      const float nmax = fmaxf(run_max[h], cmax);
      const float scale = __expf(run_max[h] - nmax);
      float csum = 0.f;
      v16h pa;
#pragma unroll
      for (int j = 0; j < 16; ++j) {
        const float p = __expf(sc[j] - nmax);
        csum += p;
        pa[j] = (_Float16)p;
      }
      csum += __shfl_xor(csum, 16, 32);
      run_sum[h] = run_sum[h] * scale + csum;
      run_max[h] = nmax;

#pragma unroll
      for (int v = 0; v < 8; ++v) {
        const float sv = __shfl(scale, v + g * 8, 32);
        acc[h][0][v] *= sv;
        acc[h][1][v] *= sv;
      }
      const _Float16* hb = wbase + (size_t)h * 16384;
#pragma unroll
      for (int t = 0; t < 2; ++t) {
        const v16h bf = load_line(hb + ((kc * 2 + t) * 32 + lane) * 16);
        acc[h][t] = wmma16(pa, bf, acc[h][t]);
      }
    }
  }

#pragma unroll
  for (int h = 0; h < 4; ++h) {
    const float inv = 1.f / run_sum[h];
#pragma unroll
    for (int t = 0; t < 2; ++t)
#pragma unroll
      for (int v = 0; v < 8; ++v) {
        const float iv = __shfl(inv, v + g * 8, 32);
        const int row = m0 + v + g * 8;
        out[((size_t)b * NM + NN + row) * HO + h * 32 + t * 16 + r] =
            eluf(acc[h][t][v] * iv);
      }
  }
}

extern "C" void kernel_launch(void* const* d_in, const int* in_sizes, int n_in,
                              void* d_out, int out_size, void* d_ws, size_t ws_size,
                              hipStream_t stream) {
  const float* fea_j    = (const float*)d_in[0];
  const int*   op_mask  = (const int*)d_in[1];
  const float* fea_m    = (const float*)d_in[2];
  const int*   mch_mask = (const int*)d_in[3];
  const float* comp_val = (const float*)d_in[4];
  const float* W_op     = (const float*)d_in[5];
  const float* a_op     = (const float*)d_in[6];
  const float* W_m      = (const float*)d_in[7];
  const float* W_edge   = (const float*)d_in[8];
  const float* a_m      = (const float*)d_in[9];
  float* out = (float*)d_out;

  char* ws = (char*)d_ws;
  size_t off = 0;
  _Float16* whmfrag = (_Float16*)(ws + off); off += (size_t)BB * 4 * 16 * 2 * 32 * 16 * 2;  // 4 MB
  float* wh1m = (float*)(ws + off); off += (size_t)BB * 4 * MM * 4;                          // 256 KB
  float* wh2m = (float*)(ws + off); off += (size_t)BB * 4 * MM * 4;                          // 256 KB
  float* whj  = (float*)(ws + off); off += (size_t)BB * NN * HO * 4;                         // 32.8 MB
  float* wh1j = (float*)(ws + off); off += (size_t)BB * 4 * NN * 4;                          // 1 MB
  float* wh2j = (float*)(ws + off); off += (size_t)BB * 4 * NN * 4;                          // 1 MB
  _Float16* wpack_op = (_Float16*)(ws + off); off += 8192 * 2;                               // 16 KB
  _Float16* wpack_m  = (_Float16*)(ws + off); off += 8192 * 2;                               // 16 KB

  hipLaunchKernelGGL(pack_w, dim3(1), dim3(256), 0, stream, W_op, W_m, wpack_op, wpack_m);
  hipLaunchKernelGGL(op_proj, dim3(NN / 16, BB), dim3(32), 0, stream,
                     fea_j, wpack_op, a_op, whj, wh1j, wh2j);
  hipLaunchKernelGGL(mch_proj, dim3(MM / 16, BB), dim3(32), 0, stream,
                     fea_m, wpack_m, a_m, whmfrag, wh1m, wh2m);
  hipLaunchKernelGGL(op_attn2, dim3(NN / 16, BB), dim3(32), 0, stream,
                     op_mask, whj, wh1j, wh2j, out);
  hipLaunchKernelGGL(mch_attn, dim3(MM / 16, BB), dim3(32), 0, stream,
                     comp_val, mch_mask, W_edge, a_m, whmfrag, wh1m, wh2m, out);
}
